// SGC_67413806678511
// MI455X (gfx1250) — matrix operand order repaired and checked by reading, not compiled, not run on hardware
//
#include <hip/hip_runtime.h>
#include <hip/hip_bf16.h>
#include <math.h>

typedef __bf16 bf16;
typedef __attribute__((ext_vector_type(4)))  __bf16 v4bf;
typedef __attribute__((ext_vector_type(16))) __bf16 v16bf;
typedef __attribute__((ext_vector_type(8)))  float  v8f;

#define DP 512      // D_PROT
#define DT 2048     // D_TEXT

// ---------------------------------------------------------------------------
// small utility kernels
// ---------------------------------------------------------------------------
__global__ void zero_u32(unsigned int* p, long n) {
    long i = (long)blockIdx.x * blockDim.x + threadIdx.x;
    if (i < n) p[i] = 0u;
}

__global__ void f32_to_bf16_k(const float* __restrict__ in, bf16* __restrict__ out, long n) {
    long i = (long)blockIdx.x * blockDim.x + threadIdx.x;
    if (i < n) out[i] = (bf16)in[i];
}

__global__ void deg_count_k(const int* __restrict__ dst, float* __restrict__ deg, int nE) {
    int i = blockIdx.x * blockDim.x + threadIdx.x;
    if (i < nE) atomicAdd(&deg[dst[i]], 1.0f);
}

__global__ void make_dinv_k(const float* __restrict__ deg, float* __restrict__ dinv, int n) {
    int i = blockIdx.x * blockDim.x + threadIdx.x;
    if (i < n) dinv[i] = rsqrtf(deg[i] + 1.0f);
}

// ---------------------------------------------------------------------------
// GraphNorm: column sums (f64 accumulation), finalize to per-column a,c, apply
// ---------------------------------------------------------------------------
__global__ void gn_sums_k(const float* __restrict__ x, double* __restrict__ csum,
                          double* __restrict__ csq, int nRows) {
    // blockDim must be 256: each thread owns 2 of the 512 columns
    int c0 = threadIdx.x, c1 = threadIdx.x + 256;
    double s0 = 0, q0 = 0, s1 = 0, q1 = 0;
    for (int r = blockIdx.x; r < nRows; r += gridDim.x) {
        const float* row = x + (size_t)r * DP;
        float v0 = row[c0], v1 = row[c1];
        s0 += v0; q0 += (double)v0 * v0;
        s1 += v1; q1 += (double)v1 * v1;
    }
    atomicAdd(&csum[c0], s0); atomicAdd(&csq[c0], q0);
    atomicAdd(&csum[c1], s1); atomicAdd(&csq[c1], q1);
}

__global__ void gn_finalize_k(const double* __restrict__ csum, const double* __restrict__ csq,
                              const float* __restrict__ w, const float* __restrict__ b,
                              const float* __restrict__ ms, float* __restrict__ ca,
                              float* __restrict__ cb, double invN) {
    int c = blockIdx.x * blockDim.x + threadIdx.x;
    if (c >= DP) return;
    double mu   = csum[c] * invN;
    double m2   = csq[c] * invN;
    double msub = mu * (double)ms[c];
    double var  = m2 - 2.0 * msub * mu + msub * msub;
    float rstd  = rsqrtf((float)var + 1e-5f);
    ca[c] = rstd * w[c];
    cb[c] = (float)(-msub) * rstd * w[c] + b[c];
}

__global__ void gn_apply_k(const float* __restrict__ x, const float* __restrict__ ca,
                           const float* __restrict__ cb, float* __restrict__ h, long n4) {
    long i4 = (long)blockIdx.x * blockDim.x + threadIdx.x;
    if (i4 >= n4) return;
    int c = (int)((i4 * 4) & (DP - 1));
    float4 xv = ((const float4*)x)[i4];
    float4 av = *(const float4*)(ca + c);
    float4 bv = *(const float4*)(cb + c);
    float4 o;
    o.x = xv.x * av.x + bv.x; o.y = xv.y * av.y + bv.y;
    o.z = xv.z * av.z + bv.z; o.w = xv.w * av.w + bv.w;
    ((float4*)h)[i4] = o;
}

// ---------------------------------------------------------------------------
// SGConv aggregation: self-loop init + edge scatter (f32 atomics, L2 resident)
// ---------------------------------------------------------------------------
__global__ void agg_init_k(const float* __restrict__ h, const float* __restrict__ dinv,
                           float* __restrict__ agg, long n4) {
    long i4 = (long)blockIdx.x * blockDim.x + threadIdx.x;
    if (i4 >= n4) return;
    int row = (int)(i4 >> 7);           // 128 float4 per 512-wide row
    float dv = dinv[row];
    float d2 = dv * dv;
    float4 hv = ((const float4*)h)[i4];
    float4 o = make_float4(hv.x * d2, hv.y * d2, hv.z * d2, hv.w * d2);
    ((float4*)agg)[i4] = o;
}

__global__ void scatter_edges_k(const float* __restrict__ h, const int* __restrict__ src,
                                const int* __restrict__ dst, const float* __restrict__ dinv,
                                float* __restrict__ agg, int nE) {
    // 2 edges per 256-thread block; 128 threads x float4 per edge row
    int e = blockIdx.x * 2 + (threadIdx.x >> 7);
    int t = threadIdx.x & 127;
    if (e >= nE) return;
    int s = src[e], d = dst[e];
    float sc = dinv[s] * dinv[d];
    float4 hv = *((const float4*)(h + (size_t)s * DP) + t);
    float* ap = agg + (size_t)d * DP + t * 4;
    atomicAdd(ap + 0, hv.x * sc);
    atomicAdd(ap + 1, hv.y * sc);
    atomicAdd(ap + 2, hv.z * sc);
    atomicAdd(ap + 3, hv.w * sc);
}

// ---------------------------------------------------------------------------
// bf16 WMMA GEMM: C[M,Nout] = epi(A[M,K] * W[Nout,K]^T + bias)
// block tile 128x64, K-step 32, 8 waves, wave tile 32x32 (2x2 WMMA 16x16x32).
// Register-staged software pipeline: global loads for tile k+1 issue while
// tile k's WMMAs execute; L2 pre-warmed with global_prefetch two tiles ahead.
// EPI: 0 = none, 1 = relu, 2 = exact gelu
// ---------------------------------------------------------------------------
template <int EPI>
__global__ __launch_bounds__(256) void gemm_bf16_k(const float* __restrict__ A,
                                                   const bf16* __restrict__ W,
                                                   const float* __restrict__ bias,
                                                   float* __restrict__ C,
                                                   int M, int K, int Nout) {
    __shared__ bf16 sA[128][40];   // 32 used + pad (80B row stride, 8B aligned)
    __shared__ bf16 sB[64][40];

    const int tid  = threadIdx.x;
    const int lane = tid & 31;
    const int wave = tid >> 5;
    const int wm0  = (wave >> 1) * 32;   // 4 waves along M
    const int wn0  = (wave & 1) * 32;    // 2 waves along N
    const int bm0  = blockIdx.y * 128;
    const int bn0  = blockIdx.x * 64;

    union Acc { v8f v; float e[8]; };
    union Frag { v16bf v; bf16 e[16]; };

    Acc acc[2][2];
#pragma unroll
    for (int tm = 0; tm < 2; ++tm)
#pragma unroll
        for (int tn = 0; tn < 2; ++tn)
#pragma unroll
            for (int r = 0; r < 8; ++r) acc[tm][tn].e[r] = 0.0f;

    const int rr = lane & 15;
    const int kh = (lane >> 4) * 8;    // A fragment K-half offset
    const int kb = (lane >> 4) * 16;   // unused; B fragments also use kh

    // per-thread staging coordinates (fixed across k-steps)
    const int arow[4] = { (tid + 0) >> 3, (tid + 256) >> 3, (tid + 512) >> 3, (tid + 768) >> 3 };
    const int acol    = (tid & 7) * 4;
    const int brow[2] = { (tid + 0) >> 3, (tid + 256) >> 3 };

    // pipeline registers
    float4 ra[4];
    v4bf   rb[2];

    auto fetch_tile = [&](int k0) {
#pragma unroll
        for (int i = 0; i < 4; ++i) {
            int gr = bm0 + arow[i];
            ra[i] = make_float4(0.f, 0.f, 0.f, 0.f);
            if (gr < M) ra[i] = *(const float4*)(A + (size_t)gr * K + k0 + acol);
        }
#pragma unroll
        for (int i = 0; i < 2; ++i) {
            rb[i] = *(const v4bf*)(W + (size_t)(bn0 + brow[i]) * K + k0 + acol);
        }
    };

    auto store_tile = [&]() {
#pragma unroll
        for (int i = 0; i < 4; ++i) {
            v4bf t;
            t.x = (bf16)ra[i].x; t.y = (bf16)ra[i].y;
            t.z = (bf16)ra[i].z; t.w = (bf16)ra[i].w;
            *(v4bf*)&sA[arow[i]][acol] = t;
        }
#pragma unroll
        for (int i = 0; i < 2; ++i) {
            *(v4bf*)&sB[brow[i]][acol] = rb[i];
        }
    };

    fetch_tile(0);

    for (int k0 = 0; k0 < K; k0 += 32) {
        store_tile();
        __syncthreads();

        if (k0 + 32 < K) {
            fetch_tile(k0 + 32);                      // overlap with WMMAs below
            if (k0 + 64 < K) {                        // pre-warm L2 two tiles out
                __builtin_prefetch(A + (size_t)(bm0 + arow[0]) * K + k0 + 64 + acol, 0, 1);
                __builtin_prefetch(W + (size_t)(bn0 + brow[0]) * K + k0 + 64 + acol, 0, 1);
            }
        }

        // build fragments per CDNA5 16-bit A(16x32)/B(32x16) VGPR layouts
        Frag a0, a1, b0, b1;
#pragma unroll
        for (int j = 0; j < 8; ++j) {
            a0.e[j]     = sA[wm0 + rr][kh + j];
            a0.e[8 + j] = sA[wm0 + rr][kh + 16 + j];
            a1.e[j]     = sA[wm0 + 16 + rr][kh + j];
            a1.e[8 + j] = sA[wm0 + 16 + rr][kh + 16 + j];
        }
#pragma unroll
        for (int j = 0; j < 8; ++j) {
            b0.e[j]     = sB[wn0 + rr][kh + j];
            b0.e[8 + j] = sB[wn0 + rr][kh + 16 + j];
            b1.e[j]     = sB[wn0 + 16 + rr][kh + j];
            b1.e[8 + j] = sB[wn0 + 16 + rr][kh + 16 + j];
        }

        acc[0][0].v = __builtin_amdgcn_wmma_f32_16x16x32_bf16(false, a0.v, false, b0.v,
                                                              (short)0, acc[0][0].v, false, false);
        acc[0][1].v = __builtin_amdgcn_wmma_f32_16x16x32_bf16(false, a0.v, false, b1.v,
                                                              (short)0, acc[0][1].v, false, false);
        acc[1][0].v = __builtin_amdgcn_wmma_f32_16x16x32_bf16(false, a1.v, false, b0.v,
                                                              (short)0, acc[1][0].v, false, false);
        acc[1][1].v = __builtin_amdgcn_wmma_f32_16x16x32_bf16(false, a1.v, false, b1.v,
                                                              (short)0, acc[1][1].v, false, false);
        __syncthreads();
    }

    // epilogue: C layout — lane&15 = column, lanes>=16 -> rows +8, vgpr r -> row +r
#pragma unroll
    for (int tm = 0; tm < 2; ++tm) {
#pragma unroll
        for (int tn = 0; tn < 2; ++tn) {
            int col   = bn0 + wn0 + tn * 16 + (lane & 15);
            int rbase = bm0 + wm0 + tm * 16 + ((lane >> 4) * 8);
            float bv = bias[col];
#pragma unroll
            for (int r = 0; r < 8; ++r) {
                int grow = rbase + r;
                if (grow < M) {
                    float v = acc[tm][tn].e[r] + bv;
                    if (EPI == 1) v = fmaxf(v, 0.0f);
                    if (EPI == 2) v = 0.5f * v * (1.0f + erff(v * 0.7071067811865476f));
                    C[(size_t)grow * Nout + col] = v;
                }
            }
        }
    }
}

// ---------------------------------------------------------------------------
// launch
// ---------------------------------------------------------------------------
extern "C" void kernel_launch(void* const* d_in, const int* in_sizes, int n_in,
                              void* d_out, int out_size, void* d_ws, size_t ws_size,
                              hipStream_t stream) {
    const float* x   = (const float*)d_in[0];
    const int*  edge = (const int*)d_in[1];
    const float* gnw = (const float*)d_in[2];
    const float* gnb = (const float*)d_in[3];
    const float* gnm = (const float*)d_in[4];
    const float* c1w = (const float*)d_in[5];
    const float* c1b = (const float*)d_in[6];
    const float* c2w = (const float*)d_in[7];
    const float* c2b = (const float*)d_in[8];
    const float* p1w = (const float*)d_in[9];
    const float* p1b = (const float*)d_in[10];
    const float* p2w = (const float*)d_in[11];
    const float* p2b = (const float*)d_in[12];
    float* out = (float*)d_out;

    const int N = in_sizes[0] / DP;   // 50000
    const int E = in_sizes[1] / 2;    // 400000
    const int* src = edge;
    const int* dst = edge + E;

    // workspace carve-up (~626 MB)
    char* base = (char*)d_ws;
    size_t off = 0;
    auto alloc = [&](size_t bytes) -> char* {
        char* p = base + off;
        off += (bytes + 255) & ~(size_t)255;
        return p;
    };
    float*  deg    = (float*)alloc((size_t)N * 4);
    float*  dinv   = (float*)alloc((size_t)N * 4);
    double* csum   = (double*)alloc(DP * 8);
    double* csq    = (double*)alloc(DP * 8);
    float*  ca     = (float*)alloc(DP * 4);
    float*  cb     = (float*)alloc(DP * 4);
    float*  bufH   = (float*)alloc((size_t)N * DP * 4);
    float*  bufAgg = (float*)alloc((size_t)N * DP * 4);
    float*  bufMid = (float*)alloc((size_t)N * DT * 4);
    bf16*   w1     = (bf16*)alloc((size_t)DP * DP * 2);
    bf16*   w2     = (bf16*)alloc((size_t)DP * DP * 2);
    bf16*   w3     = (bf16*)alloc((size_t)DT * DP * 2);
    bf16*   w4     = (bf16*)alloc((size_t)DT * DT * 2);
    (void)ws_size; (void)n_in; (void)out_size;

    const long n4 = (long)N * DP / 4;

    // weight conversions f32 -> bf16
    {
        long c1 = (long)DP * DP, c3 = (long)DT * DP, c4 = (long)DT * DT;
        f32_to_bf16_k<<<(int)((c1 + 255) / 256), 256, 0, stream>>>(c1w, w1, c1);
        f32_to_bf16_k<<<(int)((c1 + 255) / 256), 256, 0, stream>>>(c2w, w2, c1);
        f32_to_bf16_k<<<(int)((c3 + 255) / 256), 256, 0, stream>>>(p1w, w3, c3);
        f32_to_bf16_k<<<(int)((c4 + 255) / 256), 256, 0, stream>>>(p2w, w4, c4);
    }

    // degrees / dinv
    zero_u32<<<(N + 255) / 256, 256, 0, stream>>>((unsigned int*)deg, N);
    deg_count_k<<<(E + 255) / 256, 256, 0, stream>>>(dst, deg, E);
    make_dinv_k<<<(N + 255) / 256, 256, 0, stream>>>(deg, dinv, N);

    // GraphNorm
    zero_u32<<<(DP * 2 + 255) / 256, 256, 0, stream>>>((unsigned int*)csum, DP * 2);
    zero_u32<<<(DP * 2 + 255) / 256, 256, 0, stream>>>((unsigned int*)csq, DP * 2);
    gn_sums_k<<<128, 256, 0, stream>>>(x, csum, csq, N);
    gn_finalize_k<<<2, 256, 0, stream>>>(csum, csq, gnw, gnb, gnm, ca, cb, 1.0 / (double)N);
    gn_apply_k<<<(int)((n4 + 255) / 256), 256, 0, stream>>>(x, ca, cb, bufH, n4);

    dim3 g512((DP + 63) / 64, (N + 127) / 128);
    dim3 g2048((DT + 63) / 64, (N + 127) / 128);

    // conv1: agg = D^-1/2 (A+I) D^-1/2 h ; h = relu(agg @ W1^T + b1)
    agg_init_k<<<(int)((n4 + 255) / 256), 256, 0, stream>>>(bufH, dinv, bufAgg, n4);
    scatter_edges_k<<<(E + 1) / 2, 256, 0, stream>>>(bufH, src, dst, dinv, bufAgg, E);
    gemm_bf16_k<1><<<g512, 256, 0, stream>>>(bufAgg, w1, c1b, bufH, N, DP, DP);

    // conv2
    agg_init_k<<<(int)((n4 + 255) / 256), 256, 0, stream>>>(bufH, dinv, bufAgg, n4);
    scatter_edges_k<<<(E + 1) / 2, 256, 0, stream>>>(bufH, src, dst, dinv, bufAgg, E);
    gemm_bf16_k<1><<<g512, 256, 0, stream>>>(bufAgg, w2, c2b, bufH, N, DP, DP);

    // proj1 + exact gelu -> bufMid [N, 2048]
    gemm_bf16_k<2><<<g2048, 256, 0, stream>>>(bufH, w3, p1b, bufMid, N, DP, DT);

    // proj2 -> out [N, 2048]
    gemm_bf16_k<0><<<g2048, 256, 0, stream>>>(bufMid, w4, p2b, out, N, DT, DT);
}
